// MotionTextAlignmentLoss_15556371546850
// MI455X (gfx1250) — compile-verified, matrix operand order checked by reading, not run on hardware
//
#include <hip/hip_runtime.h>
#include <hip/hip_bf16.h>
#include <math.h>

// ---------------------------------------------------------------------------
// MotionTextAlignmentLoss fused kernel for gfx1250 (MI455X, wave32, WMMA)
//
//   N = 1024 + 2*1024*3 = 7168 rows, D = 512
//   sim = (E f16) x (E f16)^T accumulated in f32 via v_wmma_f32_16x16x32_f16
//   Row reductions (logsumexp denom, pos mean, margin mean) fused per tile;
//   the 205MB sim matrix is never materialized.
//
// Round-3 dataflow changes:
//   * 4 waves / block, 64 rows / block; the 16-column B panel (16KB) is staged
//     in LDS once per tile and shared by all 4 waves (4x less L2 traffic).
//   * Panels are fetched with GLOBAL_LOAD_ASYNC_TO_LDS_B128 (inline asm,
//     ASYNCcnt-tracked) and double-buffered: panel ct+1 streams in while the
//     waves compute on panel ct; sync = s_wait_asynccnt 0 + s_barrier.
//   * B fragments now come from LDS (ds_load_b128, ~short latency) instead of
//     L2, removing the per-k-step s_wait_loadcnt 0 stalls seen in round 2.
//   * LDS rows padded to 520 halves (1040B = 65*16B) -> conflict-free b128
//     reads per 16-lane half-wave, 16B alignment preserved.
// ---------------------------------------------------------------------------

#define TEMPERATURE   0.07f
#define MARGIN        0.2f
#define SHUF_OFFSET   100000
#define M_ROWS        1024
#define C_CAP         3
#define D_DIM         512
#define N_TOT         (M_ROWS + 2 * M_ROWS * C_CAP)   // 7168
#define N_COLTILES    (N_TOT / 16)                    // 448
#define N_KSTEPS      (D_DIM / 32)                    // 16
#define LDS_ROW       520                             // 512 + 8-half pad

typedef __attribute__((ext_vector_type(16))) _Float16 v16h;
typedef __attribute__((ext_vector_type(8)))  _Float16 v8h;
typedef __attribute__((ext_vector_type(8)))  float    v8f;
typedef __attribute__((ext_vector_type(4)))  float    v4f;

// --------------------------- phase 1: prep ---------------------------------
// One wave per row: normalize (x + 1e-8) / max(||.||, 1e-12), convert to f16,
// emit ids. Thread (0,0) also zeroes the loss accumulators (runs before the
// main kernel by stream ordering).
__global__ void __launch_bounds__(256)
prep_kernel(const float* __restrict__ motion,
            const float* __restrict__ text,
            const float* __restrict__ shuf,
            const int*   __restrict__ motion_ids,
            _Float16*    __restrict__ E,
            int*         __restrict__ ids,
            float*       __restrict__ accum)
{
    if (blockIdx.x == 0 && threadIdx.x == 0) { accum[0] = 0.f; accum[1] = 0.f; }

    const int wave = threadIdx.x >> 5;          // 8 waves / block
    const int lane = threadIdx.x & 31;
    const int row  = blockIdx.x * 8 + wave;
    if (row >= N_TOT) return;

    const float* src;
    int id;
    if (row < M_ROWS) {
        src = motion + (size_t)row * D_DIM;
        id  = motion_ids[row];
    } else if (row < M_ROWS + M_ROWS * C_CAP) {
        const int t = row - M_ROWS;
        src = text + (size_t)t * D_DIM;
        id  = motion_ids[t / C_CAP];
    } else {
        const int s = row - M_ROWS - M_ROWS * C_CAP;
        src = shuf + (size_t)s * D_DIM;
        id  = motion_ids[s / C_CAP] + SHUF_OFFSET;
    }

    // each lane owns 16 contiguous floats of the 512-wide row
    float v[16];
    float ss = 0.f;
    const v4f* src4 = (const v4f*)(src + lane * 16);
#pragma unroll
    for (int i = 0; i < 4; ++i) {
        v4f x = src4[i];
#pragma unroll
        for (int j = 0; j < 4; ++j) {
            float y = x[j] + 1e-8f;
            v[i * 4 + j] = y;
            ss += y * y;
        }
    }
#pragma unroll
    for (int m = 1; m < 32; m <<= 1) ss += __shfl_xor(ss, m, 32);
    const float inv = 1.f / fmaxf(sqrtf(ss), 1e-12f);

    _Float16 h[16];
#pragma unroll
    for (int i = 0; i < 16; ++i) h[i] = (_Float16)(v[i] * inv);
    v8h* dst = (v8h*)(E + (size_t)row * D_DIM + lane * 16);
    dst[0] = *(const v8h*)&h[0];
    dst[1] = *(const v8h*)&h[8];

    if (lane == 0) ids[row] = id;
}

// ------------------- phase 2: fused WMMA Gram + reduction ------------------
// 4 waves / block, each owning a 16-row tile (A panel resident in 128 VGPRs).
// B panels double-buffered in LDS via async global->LDS copies.
__global__ void __launch_bounds__(128)
loss_kernel(const _Float16* __restrict__ E,
            const int*      __restrict__ ids,
            float*          __restrict__ accum)
{
    __shared__ __align__(16) _Float16 Bpanel[2][16 * LDS_ROW];   // 2 x 16.6 KB

    const int tid  = threadIdx.x;
    const int wv   = tid >> 5;        // wave 0..3
    const int lane = tid & 31;
    const int l    = lane & 15;       // N (or M) sub-index
    const int hi   = lane >> 4;       // K half select
    const int rowBase = blockIdx.x * 64 + wv * 16;

    // --- async staging of one 16-row B panel (16KB) by the whole block ----
    // thread t copies a 128B chunk: row = t/8, chunk = t%8, via 8 async b128.
    auto issue_panel = [&](int ct, int bufsel) {
        const int r      = tid >> 3;          // 0..15
        const int cchunk = tid & 7;           // 0..7 (64 halves each)
        const _Float16* g = E + (size_t)(ct * 16 + r) * D_DIM + cchunk * 64;
        unsigned long long ga = (unsigned long long)g;
        typedef __attribute__((address_space(3))) _Float16 lds_h;
        unsigned ldsOff =
            (unsigned)(size_t)(lds_h*)&Bpanel[bufsel][r * LDS_ROW + cchunk * 64];
        asm volatile(
            "global_load_async_to_lds_b128 %0, %1, off offset:0\n"
            "global_load_async_to_lds_b128 %0, %1, off offset:16\n"
            "global_load_async_to_lds_b128 %0, %1, off offset:32\n"
            "global_load_async_to_lds_b128 %0, %1, off offset:48\n"
            "global_load_async_to_lds_b128 %0, %1, off offset:64\n"
            "global_load_async_to_lds_b128 %0, %1, off offset:80\n"
            "global_load_async_to_lds_b128 %0, %1, off offset:96\n"
            "global_load_async_to_lds_b128 %0, %1, off offset:112\n"
            :: "v"(ldsOff), "v"(ga) : "memory");
    };

    // ---- load A fragments for every k-step (rows rowBase..rowBase+15) ----
    v16h a[N_KSTEPS];
    {
        const _Float16* arow = E + (size_t)(rowBase + l) * D_DIM;
#pragma unroll
        for (int ks = 0; ks < N_KSTEPS; ++ks) {
            const int k0 = ks * 32 + hi * 8;          // K chunks {k0..k0+7, k0+16..k0+23}
            v8h c0 = *(const v8h*)(arow + k0);
            v8h c1 = *(const v8h*)(arow + k0 + 16);
            v16h f;
#pragma unroll
            for (int j = 0; j < 8; ++j) { f[j] = c0[j]; f[8 + j] = c1[j]; }
            a[ks] = f;
        }
    }

    int myRow[8], idr[8];
#pragma unroll
    for (int j = 0; j < 8; ++j) {
        myRow[j] = rowBase + hi * 8 + j;              // C layout: lanes<16 -> M 0..7
        idr[j]   = ids[myRow[j]];
    }

    float denom[8], psum[8], msum[8];
    int   npos[8];
#pragma unroll
    for (int j = 0; j < 8; ++j) { denom[j] = 0.f; psum[j] = 0.f; msum[j] = 0.f; npos[j] = 0; }

    const float invT = 1.0f / TEMPERATURE;

    // ---- double-buffer prologue: panel 0 resident, panel 1 in flight ----
    issue_panel(0, 0);
    asm volatile("s_wait_asynccnt 0x0" ::: "memory");
    __syncthreads();
    issue_panel(1, 1);

    int cur = 0;
    for (int ct = 0; ct < N_COLTILES; ++ct) {
        const int col = ct * 16 + l;
        const int idc = ids[col];
        const _Float16* bbase = &Bpanel[cur][(size_t)l * LDS_ROW];

        v8f c0acc = {0.f, 0.f, 0.f, 0.f, 0.f, 0.f, 0.f, 0.f};
        v8f c1acc = {0.f, 0.f, 0.f, 0.f, 0.f, 0.f, 0.f, 0.f};
#pragma unroll
        for (int ks = 0; ks < N_KSTEPS; ++ks) {
            const int k0 = ks * 32 + hi * 8;
            v8h b0 = *(const v8h*)(bbase + k0);        // ds_load_b128
            v8h b1 = *(const v8h*)(bbase + k0 + 16);   // ds_load_b128
            v16h bf;
#pragma unroll
            for (int j = 0; j < 8; ++j) { bf[j] = b0[j]; bf[8 + j] = b1[j]; }
            // two interleaved accumulator chains halve the WMMA RAW depth
            if (ks & 1)
                c1acc = __builtin_amdgcn_wmma_f32_16x16x32_f16(
                            false, a[ks], false, bf, (short)0, c1acc, false, false);
            else
                c0acc = __builtin_amdgcn_wmma_f32_16x16x32_f16(
                            false, a[ks], false, bf, (short)0, c0acc, false, false);
        }

        // fused per-element reduction of this 16x16 sim tile
#pragma unroll
        for (int j = 0; j < 8; ++j) {
            const float dot  = c0acc[j] + c1acc[j];
            const float s    = dot * invT;
            const bool  diag = (myRow[j] == col);
            const bool  same = (idr[j] == idc);
            if (!diag) {
                denom[j] += __expf(s);          // v_exp_f32 path
                if (same) { psum[j] += s; npos[j] += 1; }
                else      { msum[j] += fmaxf(s + MARGIN, 0.f); }
            }
        }

        // ---- rotate double buffer: wait for in-flight panel, resync, refill
        if (ct + 1 < N_COLTILES) {
            asm volatile("s_wait_asynccnt 0x0" ::: "memory");  // panel ct+1 landed
            __syncthreads();   // everyone done reading buf[cur]; asyncs visible
            if (ct + 2 < N_COLTILES)
                issue_panel(ct + 2, cur);                      // refill freed buffer
            cur ^= 1;
        }
    }

    // ---- fold the 16 lanes that share each row ----
#pragma unroll
    for (int m = 1; m < 16; m <<= 1) {
#pragma unroll
        for (int j = 0; j < 8; ++j) {
            denom[j] += __shfl_xor(denom[j], m, 32);
            psum[j]  += __shfl_xor(psum[j],  m, 32);
            msum[j]  += __shfl_xor(msum[j],  m, 32);
            npos[j]  += __shfl_xor(npos[j],  m, 32);
        }
    }

    if (l == 0) {   // lanes 0 and 16 of each wave finalize 8 rows each
        float tot = 0.f, cnt = 0.f;
#pragma unroll
        for (int j = 0; j < 8; ++j) {
            const int np = npos[j];
            const int nn = (N_TOT - 1) - np;
            if (np > 0 && nn > 0) {
                const float infonce = __logf(denom[j]) - psum[j] / (float)np;
                const float marg    = msum[j] / (float)nn;
                tot += infonce + marg;
                cnt += 1.f;
            }
        }
        atomicAdd(&accum[0], tot);
        atomicAdd(&accum[1], cnt);
    }
}

// --------------------------- phase 3: finalize -----------------------------
__global__ void finalize_kernel(const float* __restrict__ accum,
                                float* __restrict__ out)
{
    const float c = accum[1];
    out[0] = (c > 0.f) ? accum[0] / c : 0.f;
}

// ---------------------------------------------------------------------------
extern "C" void kernel_launch(void* const* d_in, const int* in_sizes, int n_in,
                              void* d_out, int out_size, void* d_ws, size_t ws_size,
                              hipStream_t stream)
{
    (void)in_sizes; (void)n_in; (void)out_size; (void)ws_size;

    const float* motion     = (const float*)d_in[0];   // [1024, 512]
    const float* text       = (const float*)d_in[1];   // [1024, 3, 512]
    const float* shuf       = (const float*)d_in[2];   // [1024, 3, 512]
    const int*   motion_ids = (const int*)d_in[3];     // [1024]
    float*       out        = (float*)d_out;

    // workspace layout
    char* ws = (char*)d_ws;
    _Float16* E     = (_Float16*)ws;                          // 7168*512*2 = 7,340,032 B
    int*      ids   = (int*)(ws + (size_t)N_TOT * D_DIM * 2); //   7168*4  =    28,672 B
    float*    accum = (float*)(ws + (size_t)N_TOT * D_DIM * 2 + (size_t)N_TOT * 4);

    // phase 1: normalize -> f16 E, ids, zero accumulators (8 rows / block)
    prep_kernel<<<N_TOT / 8, 256, 0, stream>>>(motion, text, shuf, motion_ids,
                                               E, ids, accum);
    // phase 2: 4 waves / block, 64 rows / block, LDS-staged B panels
    loss_kernel<<<N_TOT / 64, 128, 0, stream>>>(E, ids, accum);
    // phase 3: scalar
    finalize_kernel<<<1, 1, 0, stream>>>(accum, out);
}
